// HardThresholdingMultiheadAttention_55722905698895
// MI455X (gfx1250) — matrix operand order repaired and checked by reading, not compiled, not run on hardware
//
#include <hip/hip_runtime.h>
#include <hip/hip_bf16.h>
#include <math.h>

#define HIDDEN 1024
#define HEADS  16
#define HD     64
#define SEQ    2048
#define BATCH  2
#define THRESH 0.001f
#define EPSL1  1e-12f

typedef __attribute__((ext_vector_type(16))) __bf16 v16bf;
typedef __attribute__((ext_vector_type(2)))  __bf16 v2bf;
typedef __attribute__((ext_vector_type(8)))  float  v8f;
typedef __attribute__((ext_vector_type(4)))  int    v4i;
typedef __attribute__((address_space(3)))    v4i    lds_v4i;

union V16U { unsigned u[8]; uint4 q[2]; v16bf v; };

// ---- f32 -> packed bf16x2: prefer CDNA5 V_CVT_PK_BF16_F32; else native casts ----
#if __has_builtin(__builtin_amdgcn_cvt_pk_bf16_f32)
__device__ __forceinline__ unsigned packbf(float a, float b) {
    v2bf p = __builtin_amdgcn_cvt_pk_bf16_f32(a, b);
    return __builtin_bit_cast(unsigned, p);
}
#else
__device__ __forceinline__ unsigned packbf(float a, float b) {
    v2bf p = { (__bf16)a, (__bf16)b };     // let clang pick the hw bf16 cvt
    return __builtin_bit_cast(unsigned, p);
}
#endif

// ---- async global->LDS (CDNA5 GLOBAL_LOAD_ASYNC_TO_LDS_B128 + ASYNCcnt) ----
#if __has_builtin(__builtin_amdgcn_global_load_async_to_lds_b128) && \
    __has_builtin(__builtin_amdgcn_s_wait_asynccnt)
#define HAVE_ASYNC_LDS 1
#else
#define HAVE_ASYNC_LDS 0
#endif

// ---------------------------------------------------------------------------
// Kernel 1: C[4096,1024] = X @ W^T + bias, output as bf16 into workspace.
// Block = 256 threads (8 waves) covers 16 tokens x 128 cols; each wave one
// 16x16 tile; K=1024 via 32 x v_wmma_f32_16x16x32_bf16.
// ---------------------------------------------------------------------------
__global__ __launch_bounds__(256)
void proj_gemm_kernel(const float* __restrict__ X, const float* __restrict__ W,
                      const float* __restrict__ bias, unsigned short* __restrict__ out)
{
    __shared__ unsigned short sX[16 * HIDDEN];     // 32 KB: 16 tokens x 1024 feats, bf16

    const int tid = threadIdx.x;
    const int mt  = blockIdx.x >> 3;               // 0..255 token tile
    const int jg  = blockIdx.x & 7;                // 0..7 column group (128 cols)

    // ---- stage X tile (f32 -> bf16) into LDS, coalesced float4 reads ----
    {
        const int tr = tid >> 4;                   // token row 0..15
        const int fb = (tid & 15) * 64;            // 64 consecutive features
        const float4* xp = (const float4*)(X + (size_t)(mt * 16 + tr) * HIDDEN + fb);
        unsigned* sp = (unsigned*)&sX[tr * HIDDEN + fb];
#pragma unroll
        for (int i = 0; i < 16; ++i) {
            float4 f = xp[i];
            sp[i * 2 + 0] = packbf(f.x, f.y);
            sp[i * 2 + 1] = packbf(f.z, f.w);
        }
    }
    __syncthreads();

    const int wave = tid >> 5, lane = tid & 31;
    const int lm = lane & 15, lh = lane >> 4;
    const int jcol = (jg * 8 + wave) * 16 + lm;    // output column for this lane
    const float* wrow = W + (size_t)jcol * HIDDEN;

    v8f acc = {};
    for (int kc = 0; kc < 32; ++kc) {
        if (kc + 1 < 32)                           // gfx1250 global_prefetch_b8
            __builtin_prefetch(wrow + (kc + 1) * 32, 0, 0);
        V16U a, bm;
        // A: lane holds contiguous 8-short runs -> two b128 LDS loads
        a.q[0] = *(const uint4*)(sX + lm * HIDDEN + kc * 32 + lh * 8);
        a.q[1] = *(const uint4*)(sX + lm * HIDDEN + kc * 32 + 16 + lh * 8);
        // B: two 8-float runs of this W row -> 4x b128 global loads
        const float4* wp = (const float4*)(wrow + kc * 32 + lh * 8);
        float4 w0 = wp[0], w1 = wp[1], w2 = wp[4], w3 = wp[5];
        bm.u[0] = packbf(w0.x, w0.y); bm.u[1] = packbf(w0.z, w0.w);
        bm.u[2] = packbf(w1.x, w1.y); bm.u[3] = packbf(w1.z, w1.w);
        bm.u[4] = packbf(w2.x, w2.y); bm.u[5] = packbf(w2.z, w2.w);
        bm.u[6] = packbf(w3.x, w3.y); bm.u[7] = packbf(w3.z, w3.w);
        acc = __builtin_amdgcn_wmma_f32_16x16x32_bf16(false, a.v, false, bm.v,
                                                      (short)0, acc, false, false);
    }

    const float bb = bias[jcol];
#pragma unroll
    for (int r = 0; r < 8; r += 2) {               // D layout: m = r + 8*lh, n = lm
        unsigned u = packbf(acc[r] + bb, acc[r + 1] + bb);
        size_t m0 = (size_t)(mt * 16 + r + lh * 8);
        out[m0 * HIDDEN + jcol]       = (unsigned short)(u & 0xFFFFu);
        out[(m0 + 1) * HIDDEN + jcol] = (unsigned short)(u >> 16);
    }
}

// ---------------------------------------------------------------------------
// Kernel 2: fused attention for one (batch, head, 16-row q tile).
// ---------------------------------------------------------------------------
__global__ __launch_bounds__(256)
void attn_kernel(const unsigned short* __restrict__ qws,
                 const unsigned short* __restrict__ kws,
                 const unsigned short* __restrict__ vws,
                 const int* __restrict__ mask,
                 float* __restrict__ outp, float* __restrict__ wout)
{
    extern __shared__ char smem[];
    float*          sS    = (float*)smem;                           // 16x2048 f32 = 128 KB
    unsigned short* sVraw = (unsigned short*)(smem + 16 * SEQ * 4); // 64x64 bf16 = 8 KB

    const int bid = blockIdx.x;
    const int qt  = bid & 127;
    const int h   = (bid >> 7) & 15;
    const int b   = bid >> 11;

    const int tid = threadIdx.x;
    const int wave = tid >> 5, lane = tid & 31;
    const int lm = lane & 15, lh = lane >> 4;

    // ---- Q tile (16x64) into WMMA A-layout registers: 4x b128 loads ----
    V16U a0, a1;
    {
        const unsigned short* qrow = qws +
            (size_t)(b * SEQ + qt * 16 + lm) * HIDDEN + h * HD;
        a0.q[0] = *(const uint4*)(qrow + lh * 8);
        a0.q[1] = *(const uint4*)(qrow + 16 + lh * 8);
        a1.q[0] = *(const uint4*)(qrow + 32 + lh * 8);
        a1.q[1] = *(const uint4*)(qrow + 48 + lh * 8);
    }

    // ---- scores: each wave owns 16 key tiles of 16 keys (K=64 -> 2 WMMA) ----
    for (int it = 0; it < 16; ++it) {
        const int kt   = wave + it * 8;            // key tile 0..127
        const int kcol = kt * 16 + lm;             // this lane's key column
        const unsigned short* krow = kws + (size_t)(b * SEQ + kcol) * HIDDEN + h * HD;
        if (it + 1 < 16)                           // prefetch next key tile's row
            __builtin_prefetch(krow + 128 * HIDDEN, 0, 0);
        V16U b0, b1;                               // B: two 8-short runs
        b0.q[0] = *(const uint4*)(krow + lh * 8);
        b0.q[1] = *(const uint4*)(krow + 16 + lh * 8);
        b1.q[0] = *(const uint4*)(krow + 32 + lh * 8);
        b1.q[1] = *(const uint4*)(krow + 48 + lh * 8);
        v8f acc = {};
        acc = __builtin_amdgcn_wmma_f32_16x16x32_bf16(false, a0.v, false, b0.v,
                                                      (short)0, acc, false, false);
        acc = __builtin_amdgcn_wmma_f32_16x16x32_bf16(false, a1.v, false, b1.v,
                                                      (short)0, acc, false, false);
        const int km = mask[b * SEQ + kcol];       // column-uniform per lane
#pragma unroll
        for (int r = 0; r < 8; ++r) {
            float s = acc[r] * 0.125f;             // 1/sqrt(64)
            if (km > 0) s = -3.0e38f;              // key_padding_mask -> -inf
            sS[(r + lh * 8) * SEQ + kcol] = s;
        }
    }
    __syncthreads();

    // ---- softmax + hard threshold + L1 renormalize; write w to HBM ----
    {
        const int r = tid >> 4;                    // row 0..15
        const int j = tid & 15;                    // 16 lanes cooperate per row
        float* row = sS + r * SEQ;

        float mx = -3.4e38f;
        for (int t = j; t < SEQ; t += 16) mx = fmaxf(mx, row[t]);
#pragma unroll
        for (int o = 8; o; o >>= 1) mx = fmaxf(mx, __shfl_xor(mx, o, 16));

        float sum = 0.f;
        for (int t = j; t < SEQ; t += 16) {
            float e = __expf(row[t] - mx);
            row[t] = e; sum += e;
        }
#pragma unroll
        for (int o = 8; o; o >>= 1) sum += __shfl_xor(sum, o, 16);
        const float inv = 1.0f / sum;

        float l1 = 0.f;
        for (int t = j; t < SEQ; t += 16) {
            float p = row[t] * inv;
            row[t] = p;
            l1 += (p >= THRESH) ? p : 0.f;         // reference: w < 0.001 -> 0
        }
#pragma unroll
        for (int o = 8; o; o >>= 1) l1 += __shfl_xor(l1, o, 16);
        const float invl1 = 1.0f / fmaxf(l1, EPSL1);

        float* wrow = wout + ((size_t)((b * HEADS + h) * SEQ + qt * 16 + r)) * SEQ;
        for (int t = j; t < SEQ; t += 16) {
            float p = row[t];
            float wv = (p >= THRESH) ? p * invl1 : 0.f;
            row[t] = wv;                           // kept for the w@V GEMM
            wrow[t] = wv;                          // the 537 MB streaming store
        }
    }
    __syncthreads();

    // ---- out = w @ V : 16x64 tile, K=2048 in 32 chunks of 64 keys ----
    v8f oacc = {};
    const int ntile = wave & 3;                    // waves 0..3 own the 4 feature tiles
    for (int kc = 0; kc < 32; ++kc) {
        {   // stage V chunk row-major: sVraw[key 0..63][feat 0..63]
            const int kr = tid >> 2;               // key row 0..63
            const int fg = (tid & 3) * 16;         // 16 consecutive features (32 B)
            const unsigned short* src = vws +
                (size_t)(b * SEQ + kc * 64 + kr) * HIDDEN + h * HD + fg;
            unsigned short* dst = sVraw + kr * 64 + fg;
            if (kc + 1 < 32)                       // prefetch next chunk's row
                __builtin_prefetch(src + 64 * HIDDEN, 0, 0);
#if HAVE_ASYNC_LDS
            __builtin_amdgcn_global_load_async_to_lds_b128(
                (v4i*)src, (lds_v4i*)dst, 0, 0);
            __builtin_amdgcn_global_load_async_to_lds_b128(
                (v4i*)(src + 8), (lds_v4i*)(dst + 8), 0, 0);
            __builtin_amdgcn_s_wait_asynccnt(0);
#else
            uint4 v0 = ((const uint4*)src)[0];
            uint4 v1 = ((const uint4*)src)[1];
            ((uint4*)dst)[0] = v0;
            ((uint4*)dst)[1] = v1;
#endif
        }
        __syncthreads();
        if (wave < 4) {                            // wave-uniform: EXEC stays all-1s
#pragma unroll
            for (int st = 0; st < 2; ++st) {
                V16U A, B;
                // A from final w (f32 in LDS): 8 consecutive floats per half
                const float* sp = sS + lm * SEQ + kc * 64 + st * 32 + lh * 8;
                float4 x0 = *(const float4*)(sp);
                float4 x1 = *(const float4*)(sp + 4);
                float4 x2 = *(const float4*)(sp + 16);
                float4 x3 = *(const float4*)(sp + 20);
                A.u[0] = packbf(x0.x, x0.y); A.u[1] = packbf(x0.z, x0.w);
                A.u[2] = packbf(x1.x, x1.y); A.u[3] = packbf(x1.z, x1.w);
                A.u[4] = packbf(x2.x, x2.y); A.u[5] = packbf(x2.z, x2.w);
                A.u[6] = packbf(x3.x, x3.y); A.u[7] = packbf(x3.z, x3.w);
#pragma unroll
                for (int i = 0; i < 8; ++i) {
                    // B(k=key, n=feat): two b16 reads from row-major sVraw, pack
                    int kl = st * 32 + lh * 8 + (i & 3) * 2 + (i >> 2) * 16;
                    int nf = ntile * 16 + lm;
                    unsigned lo = sVraw[kl * 64 + nf];
                    unsigned hi = sVraw[(kl + 1) * 64 + nf];
                    B.u[i] = lo | (hi << 16);
                }
                oacc = __builtin_amdgcn_wmma_f32_16x16x32_bf16(false, A.v, false, B.v,
                                                               (short)0, oacc, false, false);
            }
        }
        __syncthreads();
    }
    if (wave < 4) {
#pragma unroll
        for (int r = 0; r < 8; ++r) {
            int m  = qt * 16 + r + lh * 8;
            int nf = ntile * 16 + lm;
            outp[(size_t)(b * SEQ + m) * HIDDEN + h * HD + nf] = oacc[r];
        }
    }
}

// ---------------------------------------------------------------------------
extern "C" void kernel_launch(void* const* d_in, const int* in_sizes, int n_in,
                              void* d_out, int out_size, void* d_ws, size_t ws_size,
                              hipStream_t stream)
{
    (void)in_sizes; (void)n_in; (void)out_size; (void)ws_size;

    const float* query = (const float*)d_in[0];
    const float* key   = (const float*)d_in[1];
    const float* value = (const float*)d_in[2];
    const int*   mask  = (const int*)d_in[3];
    const float* Wq = (const float*)d_in[4];
    const float* bq = (const float*)d_in[5];
    const float* Wk = (const float*)d_in[6];
    const float* bk = (const float*)d_in[7];
    const float* Wv = (const float*)d_in[8];
    const float* bv = (const float*)d_in[9];

    // workspace: projected Q/K/V as bf16, 8 MB each (24 MB total)
    unsigned short* qws = (unsigned short*)d_ws;
    unsigned short* kws = qws + (size_t)BATCH * SEQ * HIDDEN;
    unsigned short* vws = kws + (size_t)BATCH * SEQ * HIDDEN;

    float* outp = (float*)d_out;                              // [2,2048,1024]
    float* wout = outp + (size_t)BATCH * SEQ * HIDDEN;        // [2,16,2048,2048]

    dim3 blk(256);
    proj_gemm_kernel<<<dim3(2048), blk, 0, stream>>>(query, Wq, bq, qws);
    proj_gemm_kernel<<<dim3(2048), blk, 0, stream>>>(key,   Wk, bk, kws);
    proj_gemm_kernel<<<dim3(2048), blk, 0, stream>>>(value, Wv, bv, vws);

    const size_t shmem = (size_t)16 * SEQ * sizeof(float)          // scores, 128 KB
                       + (size_t)64 * 64 * sizeof(unsigned short); // V tile, 8 KB
    attn_kernel<<<dim3(BATCH * HEADS * (SEQ / 16)), blk, shmem, stream>>>(
        qws, kws, vws, mask, outp, wout);
}
